// MambaFusion_68659347194378
// MI455X (gfx1250) — compile-verified, matrix operand order checked by reading, not run on hardware
//
#include <hip/hip_runtime.h>
#include <hip/hip_bf16.h>

// ---------------------------------------------------------------------------
// MambaFusion forward for gfx1250 (MI455X, wave32, WMMA).
// - All dense projections go through a register-blocked WMMA f16 GEMM
//   (f32 accumulate): each wave owns a 32x32 C block = 4 accumulators,
//   4 v_wmma_f32_16x16x32_f16 per K-step with 2xA / 2xB fragment reuse.
// - Tiles are staged to LDS with gfx1250 async copies
//   (global_load_async_to_lds_b128, ASYNCcnt), double-buffered.
// - SSM scan keeps its 16-wide state in VGPRs; LayerNorm is one wave32 per
//   row with shuffle reductions; tokens stay in (B,N,C)==NHWC everywhere.
// ---------------------------------------------------------------------------

typedef __attribute__((ext_vector_type(16))) _Float16 v16h;
typedef __attribute__((ext_vector_type(8)))  float    v8f;

#define DSN   16   // SSM state dim
#define KCONV 4    // causal dwconv taps

#if defined(__AMDGCN__) && __has_builtin(__builtin_amdgcn_global_load_async_to_lds_b128) && __has_builtin(__builtin_amdgcn_s_wait_asynccnt)
#define HAVE_ASYNC_LDS 1
#else
#define HAVE_ASYNC_LDS 0
#endif

#if HAVE_ASYNC_LDS
// Builtin expects pointers to int4 vectors (per hipcc diagnostic):
//   param: '__attribute__((__vector_size__(4*sizeof(int)))) int *'
typedef int v4i_raw __attribute__((vector_size(16)));
typedef __attribute__((address_space(1))) v4i_raw gv4i_t;   // global int4
typedef __attribute__((address_space(3))) v4i_raw lv4i_t;   // LDS int4
#endif

__device__ __forceinline__ float dev_sigmoid(float v){ return 1.f/(1.f+__expf(-v)); }
__device__ __forceinline__ float dev_silu(float v){ return v*dev_sigmoid(v); }
__device__ __forceinline__ float dev_softplus(float v){ return (v>20.f)? v : __logf(1.f+__expf(v)); }
__device__ __forceinline__ float dev_act(float v, int act){
  if(act==1) return dev_silu(v);
  if(act==2) return dev_sigmoid(v);
  if(act==3) return dev_softplus(v);
  return v;
}

// ---------------------------------------------------------------------------
// WMMA GEMM: C[M,N] = act(A[M,K] @ W[K,N] + bias[N])
// Block: 256 threads = 8 wave32 waves arranged 2x4; block tile 64x128x32.
// Each wave: 32x32 C block -> acc00..acc11, 4 WMMAs per K-step.
// LDS holds raw f32 tiles (async-copied); f32->f16 conversion happens during
// fragment build and co-executes with the WMMAs.
// Fragment layouts per CDNA5 ISA 7.12.2:
//   A 16x32 f16: lane L row M=L%16; v<4: K=(L<16?0:8)+2v+h, v>=4: +16.
//   C 16x16 f32: VGPR r -> M=r+(L<16?0:8), N=L%16.
// ---------------------------------------------------------------------------
#define BM 64
#define BN 128
#define BK 32
#define LDA (BK+4)   // f32 stride, keeps rows 16B-aligned for b128 async
#define LDB (BN+4)

__global__ void __launch_bounds__(256)
k_gemm_wmma(const float* __restrict__ A, const float* __restrict__ W,
            const float* __restrict__ bias, float* __restrict__ C,
            int M, int N, int K, int act){
  __shared__ float As[2][BM*LDA];
  __shared__ float Bs[2][BK*LDB];
  const int tid  = threadIdx.x;
  const int wid  = tid >> 5;
  const int lane = tid & 31;
  const int wm = wid & 1;     // 0..1 -> 32-row slab
  const int wn = wid >> 1;    // 0..3 -> 32-col slab
  const int m0 = blockIdx.y*BM, n0 = blockIdx.x*BN;

  v8f acc00={0.f,0.f,0.f,0.f,0.f,0.f,0.f,0.f};
  v8f acc01=acc00, acc10=acc00, acc11=acc00;

  // ---- tile staging (async b128 -> LDS when interior & aligned) ----
  auto stage = [&](int kt, int buf){
    bool fullK = (kt + BK <= K);
#if HAVE_ASYNC_LDS
    if(fullK && (m0 + BM <= M) && ((K & 3) == 0)){
      for(int i=tid; i<BM*(BK/4); i+=256){
        int r = i/(BK/4), c4 = (i%(BK/4))*4;
        const float* gp = A + (size_t)(m0+r)*K + kt + c4;
        float* lp = &As[buf][r*LDA + c4];
        __builtin_amdgcn_global_load_async_to_lds_b128((gv4i_t*)gp, (lv4i_t*)lp, 0, 0);
      }
    } else
#endif
    {
      for(int i=tid; i<BM*BK; i+=256){
        int r = i>>5, kk = i&(BK-1);
        int gm = m0+r, gk = kt+kk;
        As[buf][r*LDA+kk] = (gm<M && gk<K) ? A[(size_t)gm*K+gk] : 0.f;
      }
    }
#if HAVE_ASYNC_LDS
    if(fullK && (n0 + BN <= N) && ((N & 3) == 0)){
      for(int i=tid; i<BK*(BN/4); i+=256){
        int r = i/(BN/4), c4 = (i%(BN/4))*4;
        const float* gp = W + (size_t)(kt+r)*N + n0 + c4;
        float* lp = &Bs[buf][r*LDB + c4];
        __builtin_amdgcn_global_load_async_to_lds_b128((gv4i_t*)gp, (lv4i_t*)lp, 0, 0);
      }
    } else
#endif
    {
      for(int i=tid; i<BK*BN; i+=256){
        int r = i>>7, nn = i&(BN-1);
        int gk = kt+r, gn = n0+nn;
        Bs[buf][r*LDB+nn] = (gk<K && gn<N) ? W[(size_t)gk*N+gn] : 0.f;
      }
    }
  };

  stage(0, 0);
#if HAVE_ASYNC_LDS
  __builtin_amdgcn_s_wait_asynccnt(0);
#endif
  __syncthreads();

  int buf = 0;
  for(int kt=0; kt<K; kt+=BK){
    if(kt + BK < K) stage(kt + BK, buf^1);   // prefetch next tile (other buffer)

    // ---- build fragments (f32 LDS -> f16 regs) ----
    v16h af0, af1, bf0, bf1;
    const int Mi  = lane & 15;
    const int hiK = (lane>>4) ? 8 : 0;
    #pragma unroll
    for(int e=0;e<16;e++){
      int v = e>>1, hh = e&1;
      int k = (v<4) ? (hiK + 2*v + hh) : (hiK + 16 + 2*(v-4) + hh);
      af0[e] = (_Float16)As[buf][(wm*32 +      Mi)*LDA + k];
      af1[e] = (_Float16)As[buf][(wm*32 + 16 + Mi)*LDA + k];
      bf0[e] = (_Float16)Bs[buf][k*LDB + wn*32 +      Mi];
      bf1[e] = (_Float16)Bs[buf][k*LDB + wn*32 + 16 + Mi];
    }
    acc00 = __builtin_amdgcn_wmma_f32_16x16x32_f16(false, af0, false, bf0, (short)0, acc00, false, false);
    acc01 = __builtin_amdgcn_wmma_f32_16x16x32_f16(false, af0, false, bf1, (short)0, acc01, false, false);
    acc10 = __builtin_amdgcn_wmma_f32_16x16x32_f16(false, af1, false, bf0, (short)0, acc10, false, false);
    acc11 = __builtin_amdgcn_wmma_f32_16x16x32_f16(false, af1, false, bf1, (short)0, acc11, false, false);

#if HAVE_ASYNC_LDS
    __builtin_amdgcn_s_wait_asynccnt(0);
#endif
    __syncthreads();
    buf ^= 1;
  }

  // ---- epilogue: fused bias + activation ----
  const int Ni   = lane & 15;
  const int mAdd = (lane>>4) ? 8 : 0;
  auto store_tile = [&](const v8f& acc, int mOfs, int nOfs){
    #pragma unroll
    for(int r=0;r<8;r++){
      int row = m0 + mOfs + r + mAdd;
      int col = n0 + nOfs + Ni;
      if(row<M && col<N){
        float v = acc[r] + (bias ? bias[col] : 0.f);
        C[(size_t)row*N+col] = dev_act(v, act);
      }
    }
  };
  store_tile(acc00, wm*32,      wn*32);
  store_tile(acc01, wm*32,      wn*32 + 16);
  store_tile(acc10, wm*32 + 16, wn*32);
  store_tile(acc11, wm*32 + 16, wn*32 + 16);
}

// ---------------------------------------------------------------------------
// LayerNorm: one wave32 per row, shuffle reduction.
// ---------------------------------------------------------------------------
__global__ void k_layernorm(const float* __restrict__ x, const float* __restrict__ g,
                            const float* __restrict__ b, float* __restrict__ y,
                            int rows, int C){
  int row  = blockIdx.x*(blockDim.x>>5) + (threadIdx.x>>5);
  int lane = threadIdx.x & 31;
  if(row >= rows) return;
  const float* xr = x + (size_t)row*C;
  float s=0.f, s2=0.f;
  for(int c=lane;c<C;c+=32){ float v=xr[c]; s+=v; s2+=v*v; }
  for(int o=16;o>0;o>>=1){ s += __shfl_down(s,o,32); s2 += __shfl_down(s2,o,32); }
  s  = __shfl(s,0,32);
  s2 = __shfl(s2,0,32);
  float m = s/C, var = s2/C - m*m;
  float inv = rsqrtf(var + 1e-5f);
  float* yr = y + (size_t)row*C;
  for(int c=lane;c<C;c+=32) yr[c] = (xr[c]-m)*inv*g[c] + b[c];
}

// ---------------------------------------------------------------------------
// Elementwise / data-movement helpers
// ---------------------------------------------------------------------------
__global__ void k_slice(const float* __restrict__ src, float* __restrict__ dst,
                        int rows, int srcC, int off, int cnt){
  size_t i = (size_t)blockIdx.x*blockDim.x + threadIdx.x;
  size_t tot = (size_t)rows*cnt;
  if(i>=tot) return;
  int c = (int)(i % cnt); size_t r = i / cnt;
  dst[i] = src[r*srcC + off + c];
}
__global__ void k_swap_half(const float* __restrict__ rn, const float* __restrict__ inn,
                            float* __restrict__ rsw, float* __restrict__ isw,
                            size_t tot, int C){
  size_t i = (size_t)blockIdx.x*blockDim.x + threadIdx.x;
  if(i>=tot) return;
  bool lo = (int)(i % C) < (C>>1);
  rsw[i] = lo ? inn[i] : rn[i];
  isw[i] = lo ? rn[i]  : inn[i];
}
__global__ void k_add(const float* __restrict__ a, const float* __restrict__ b,
                      float* __restrict__ o, size_t n){
  size_t i = (size_t)blockIdx.x*blockDim.x + threadIdx.x;
  if(i<n) o[i] = a[i] + b[i];
}
__global__ void k_gate_mul(const float* __restrict__ y, const float* __restrict__ z,
                           float* __restrict__ o, size_t n){
  size_t i = (size_t)blockIdx.x*blockDim.x + threadIdx.x;
  if(i<n) o[i] = y[i]*dev_silu(z[i]);
}
__global__ void k_copy(const float* __restrict__ a, float* __restrict__ o, size_t n){
  size_t i = (size_t)blockIdx.x*blockDim.x + threadIdx.x;
  if(i<n) o[i] = a[i];
}
__global__ void k_fill(float* __restrict__ o, float v, size_t n){
  size_t i = (size_t)blockIdx.x*blockDim.x + threadIdx.x;
  if(i<n) o[i] = v;
}

// ---------------------------------------------------------------------------
// Causal depthwise conv1d over tokens + SiLU.  x,y: (B,N,di)
// ---------------------------------------------------------------------------
__global__ void k_dwconv1d_silu(const float* __restrict__ x, const float* __restrict__ w,
                                const float* __restrict__ bias, float* __restrict__ y,
                                int Bb, int N, int di){
  size_t i = (size_t)blockIdx.x*blockDim.x + threadIdx.x;
  size_t tot = (size_t)Bb*N*di;
  if(i>=tot) return;
  int d = (int)(i % di);
  size_t row = i / di;
  int t = (int)(row % N), b = (int)(row / N);
  float acc = bias[d];
  #pragma unroll
  for(int k=0;k<KCONV;k++){
    int tt = t - (KCONV-1) + k;
    if(tt>=0) acc += x[((size_t)b*N+tt)*di + d] * w[d*KCONV + k];
  }
  y[i] = dev_silu(acc);
}

// ---------------------------------------------------------------------------
// SSM scan. One lane per (b, channel); h[16] state lives in VGPRs.
// delta is pre-softplus'ed by the GEMM epilogue (act=3).
// ---------------------------------------------------------------------------
__global__ void k_ssm_scan(const float* __restrict__ x, const float* __restrict__ delta,
                           const float* __restrict__ Bp, const float* __restrict__ Cp,
                           const float* __restrict__ A_log, const float* __restrict__ Dp,
                           float* __restrict__ y, int Bb, int N, int di){
  int d = blockIdx.x*blockDim.x + threadIdx.x;
  int b = blockIdx.y;
  if(d>=di) return;
  float A[DSN], h[DSN];
  #pragma unroll
  for(int s=0;s<DSN;s++){ A[s] = -__expf(A_log[(size_t)d*DSN+s]); h[s]=0.f; }
  float Dd = Dp[d];
  for(int t=0;t<N;t++){
    size_t row = (size_t)b*N + t;
    float xv = x[row*di+d], dl = delta[row*di+d];
    const float* Bt = Bp + row*DSN;
    const float* Ct = Cp + row*DSN;
    float accv = 0.f;
    #pragma unroll
    for(int s=0;s<DSN;s++){
      h[s] = __expf(dl*A[s])*h[s] + dl*Bt[s]*xv;
      accv += h[s]*Ct[s];
    }
    y[row*di+d] = accv + Dd*xv;
  }
}

// ---------------------------------------------------------------------------
// SE block: mean over tokens -> relu(a@w1+b1) -> sigmoid(.@w2+b2) -> scale.
// ---------------------------------------------------------------------------
__global__ void k_se(const float* __restrict__ x, const float* __restrict__ w1,
                     const float* __restrict__ b1, const float* __restrict__ w2,
                     const float* __restrict__ b2, float* __restrict__ xo,
                     int N, int C, int H){
  extern __shared__ float sm[];
  float* a = sm; float* hid = sm + C;
  int b = blockIdx.x, tid = threadIdx.x;
  const float* xb = x + (size_t)b*N*C;
  float* xob = xo + (size_t)b*N*C;
  if(tid<C){ float s=0.f; for(int t=0;t<N;t++) s += xb[(size_t)t*C+tid]; a[tid]=s/N; }
  __syncthreads();
  if(tid<H){ float s=b1[tid]; for(int c=0;c<C;c++) s += a[c]*w1[c*H+tid]; hid[tid]=fmaxf(s,0.f); }
  __syncthreads();
  if(tid<C){
    float s=b2[tid]; for(int j=0;j<H;j++) s += hid[j]*w2[j*C+tid];
    float gt = dev_sigmoid(s);
    for(int t=0;t<N;t++) xob[(size_t)t*C+tid] = xb[(size_t)t*C+tid]*gt;
  }
}

// ---------------------------------------------------------------------------
// 3x3 conv over NHWC with channel-concat input [x1 | x2], optional residual,
// zero or reflect padding.
// ---------------------------------------------------------------------------
__global__ void k_conv3x3(const float* __restrict__ x1, const float* __restrict__ x2,
                          const float* __restrict__ w, const float* __restrict__ bias,
                          const float* __restrict__ resid, float* __restrict__ y,
                          int Bb, int Hh, int Ww, int C1, int C2, int Cout, int reflect){
  size_t i = (size_t)blockIdx.x*blockDim.x + threadIdx.x;
  size_t tot = (size_t)Bb*Hh*Ww*Cout;
  if(i>=tot) return;
  int oc = (int)(i % Cout);
  int px = (int)((i / Cout) % Ww);
  int py = (int)((i / ((size_t)Cout*Ww)) % Hh);
  int b  = (int)(i / ((size_t)Cout*Ww*Hh));
  int Cin = C1 + C2;
  float accv = bias[oc];
  for(int ky=0;ky<3;ky++) for(int kx=0;kx<3;kx++){
    int yy = py+ky-1, xx = px+kx-1;
    if(reflect){
      if(yy<0) yy=-yy; if(yy>=Hh) yy=2*Hh-2-yy;
      if(xx<0) xx=-xx; if(xx>=Ww) xx=2*Ww-2-xx;
    } else if(yy<0||yy>=Hh||xx<0||xx>=Ww) continue;
    const float* px1 = x1 + ((size_t)(b*Hh+yy)*Ww+xx)*C1;
    for(int ic=0;ic<C1;ic++)
      accv += px1[ic] * w[((size_t)(oc*Cin+ic)*3+ky)*3+kx];
    if(C2>0){
      const float* px2 = x2 + ((size_t)(b*Hh+yy)*Ww+xx)*C2;
      for(int ic=0;ic<C2;ic++)
        accv += px2[ic] * w[((size_t)(oc*Cin+C1+ic)*3+ky)*3+kx];
    }
  }
  y[i] = accv + (resid ? resid[i] : 0.f);
}

// Depthwise 3x3 NHWC; y = conv(x) + x (self residual, m3 block)
__global__ void k_dwconv3x3(const float* __restrict__ x, const float* __restrict__ w,
                            const float* __restrict__ bias, float* __restrict__ y,
                            int Bb, int Hh, int Ww, int C){
  size_t i = (size_t)blockIdx.x*blockDim.x + threadIdx.x;
  size_t tot = (size_t)Bb*Hh*Ww*C;
  if(i>=tot) return;
  int c  = (int)(i % C);
  int px = (int)((i / C) % Ww);
  int py = (int)((i / ((size_t)C*Ww)) % Hh);
  int b  = (int)(i / ((size_t)C*Ww*Hh));
  float accv = bias[c];
  for(int ky=0;ky<3;ky++) for(int kx=0;kx<3;kx++){
    int yy=py+ky-1, xx=px+kx-1;
    if(yy<0||yy>=Hh||xx<0||xx>=Ww) continue;
    accv += x[((size_t)(b*Hh+yy)*Ww+xx)*C + c] * w[c*9 + ky*3 + kx];
  }
  y[i] = accv + x[i];
}

// softmax over 2 attention channels + blend
__global__ void k_softmax2_blend(const float* __restrict__ att, const float* __restrict__ Ir,
                                 const float* __restrict__ Ii, float* __restrict__ ff,
                                 size_t rows, int C){
  size_t i = (size_t)blockIdx.x*blockDim.x + threadIdx.x;
  size_t tot = rows*C;
  if(i>=tot) return;
  size_t r = i / C;
  float a0 = att[r*2], a1 = att[r*2+1];
  float m = fmaxf(a0,a1);
  float e0 = __expf(a0-m), e1 = __expf(a1-m);
  float inv = 1.f/(e0+e1);
  ff[i] = (e0*Ir[i] + e1*Ii[i])*inv;
}

// 1x1 conv (skip): w layout (oc, ic)
__global__ void k_conv1x1(const float* __restrict__ x, const float* __restrict__ w,
                          const float* __restrict__ bias, float* __restrict__ y,
                          size_t rows, int C){
  size_t i = (size_t)blockIdx.x*blockDim.x + threadIdx.x;
  size_t tot = rows*C;
  if(i>=tot) return;
  int oc = (int)(i % C); size_t r = i / C;
  float accv = bias[oc];
  const float* xr = x + r*C;
  for(int ic=0;ic<C;ic++) accv += xr[ic]*w[(size_t)oc*C+ic];
  y[i] = accv;
}

// ===========================================================================
// Host side
// ===========================================================================
struct WS {
  char* p; size_t off, cap;
  float* alloc(size_t n){
    float* r = (float*)(p + off);
    off += ((n*sizeof(float) + 255) & ~(size_t)255);
    return r;
  }
};
struct Cur { void* const* din; int i; const float* next(){ return (const float*)din[i++]; } };

struct PMamba { const float *in_w,*conv_w,*conv_b,*x_w,*dt_w,*dt_b,*A_log,*D,*out_w; };
struct PCross { const float *in_w,*conv_w,*conv_b,*xdt_w,*dt_w,*dt_b,*B_w,*C_w,*A_log,*D,*out_w; };
struct PSE    { const float *w1,*b1,*w2,*b2; };
struct PTS    { const float *ln1_g,*ln1_b,*ln2_g,*ln2_b; PMamba m_rgb,m_ir; PSE se_rgb,se_ir; };
struct PM3    { const float *ln0_g,*ln0_b,*ln1_g,*ln1_b,*ln2_g,*ln2_b; PCross mamba; const float *dw_w,*dw_b; };
struct PLvl   { PTS ts1,ts2; const float *sf1_w,*sf1_b,*sf2_w,*sf2_b,*la_w,*la_b,*skip_w,*skip_b,*pe_g,*pe_b; PM3 m3[3]; };

static void load_mamba(Cur& c, PMamba& m){ m.in_w=c.next(); m.conv_w=c.next(); m.conv_b=c.next(); m.x_w=c.next(); m.dt_w=c.next(); m.dt_b=c.next(); m.A_log=c.next(); m.D=c.next(); m.out_w=c.next(); }
static void load_cross(Cur& c, PCross& m){ m.in_w=c.next(); m.conv_w=c.next(); m.conv_b=c.next(); m.xdt_w=c.next(); m.dt_w=c.next(); m.dt_b=c.next(); m.B_w=c.next(); m.C_w=c.next(); m.A_log=c.next(); m.D=c.next(); m.out_w=c.next(); }
static void load_se(Cur& c, PSE& s){ s.w1=c.next(); s.b1=c.next(); s.w2=c.next(); s.b2=c.next(); }
static void load_ts(Cur& c, PTS& t){ t.ln1_g=c.next(); t.ln1_b=c.next(); t.ln2_g=c.next(); t.ln2_b=c.next(); load_mamba(c,t.m_rgb); load_mamba(c,t.m_ir); load_se(c,t.se_rgb); load_se(c,t.se_ir); }
static void load_m3(Cur& c, PM3& m){ m.ln0_g=c.next(); m.ln0_b=c.next(); m.ln1_g=c.next(); m.ln1_b=c.next(); m.ln2_g=c.next(); m.ln2_b=c.next(); load_cross(c,m.mamba); m.dw_w=c.next(); m.dw_b=c.next(); }
static void load_level(Cur& c, PLvl& L){
  load_ts(c,L.ts1); load_ts(c,L.ts2);
  L.sf1_w=c.next(); L.sf1_b=c.next(); L.sf2_w=c.next(); L.sf2_b=c.next();
  L.la_w=c.next();  L.la_b=c.next();  L.skip_w=c.next(); L.skip_b=c.next();
  L.pe_g=c.next();  L.pe_b=c.next();
  for(int j=0;j<3;j++) load_m3(c,L.m3[j]);
}

static inline unsigned gblk(size_t n){ return (unsigned)((n + 255)/256); }

static void run_gemm(hipStream_t st, const float* A, const float* W, const float* bias,
                     float* C, int M, int N, int K, int act){
  dim3 g((N+BN-1)/BN, (M+BM-1)/BM), b(256);
  k_gemm_wmma<<<g,b,0,st>>>(A,W,bias,C,M,N,K,act);
}
static void run_ln(hipStream_t st, const float* x, const float* g, const float* b,
                   float* y, int rows, int C){
  k_layernorm<<<(rows+7)/8, 256, 0, st>>>(x,g,b,y,rows,C);
}
static void run_slice(hipStream_t st, const float* s, float* d, int rows, int sc, int off, int cnt){
  k_slice<<<gblk((size_t)rows*cnt),256,0,st>>>(s,d,rows,sc,off,cnt);
}

static void run_mamba(hipStream_t st, WS& ws, const PMamba& p, const float* x,
                      float* out, int Bb, int N, int d){
  int di = 2*d, dtr = (d+15)/16, M = Bb*N;
  size_t mark = ws.off;
  float* xz = ws.alloc((size_t)M*2*di);
  run_gemm(st, x, p.in_w, nullptr, xz, M, 2*di, d, 0);
  float* xi = ws.alloc((size_t)M*di); float* z = ws.alloc((size_t)M*di);
  run_slice(st, xz, xi, M, 2*di, 0, di);
  run_slice(st, xz, z,  M, 2*di, di, di);
  float* xc = ws.alloc((size_t)M*di);
  k_dwconv1d_silu<<<gblk((size_t)M*di),256,0,st>>>(xi,p.conv_w,p.conv_b,xc,Bb,N,di);
  int xwN = dtr + 2*DSN;
  float* xdbl = ws.alloc((size_t)M*xwN);
  run_gemm(st, xc, p.x_w, nullptr, xdbl, M, xwN, di, 0);
  float* dtc = ws.alloc((size_t)M*dtr);
  float* Bp  = ws.alloc((size_t)M*DSN);
  float* Cp  = ws.alloc((size_t)M*DSN);
  run_slice(st, xdbl, dtc, M, xwN, 0, dtr);
  run_slice(st, xdbl, Bp,  M, xwN, dtr, DSN);
  run_slice(st, xdbl, Cp,  M, xwN, dtr+DSN, DSN);
  float* delta = ws.alloc((size_t)M*di);
  run_gemm(st, dtc, p.dt_w, p.dt_b, delta, M, di, dtr, 3);   // fused softplus
  float* yv = ws.alloc((size_t)M*di);
  k_ssm_scan<<<dim3((di+127)/128,(unsigned)Bb),128,0,st>>>(xc,delta,Bp,Cp,p.A_log,p.D,yv,Bb,N,di);
  float* gt = ws.alloc((size_t)M*di);
  k_gate_mul<<<gblk((size_t)M*di),256,0,st>>>(yv,z,gt,(size_t)M*di);
  run_gemm(st, gt, p.out_w, nullptr, out, M, d, di, 0);
  ws.off = mark;
}

static void run_cross(hipStream_t st, WS& ws, const PCross& p, const float* f,
                      const float* r, const float* i, float* out, int Bb, int N, int d){
  int di = 2*d, dtr = (d+15)/16, M = Bb*N;
  size_t mark = ws.off;
  float* xz = ws.alloc((size_t)M*2*di);
  run_gemm(st, f, p.in_w, nullptr, xz, M, 2*di, d, 0);
  float* xi = ws.alloc((size_t)M*di); float* z = ws.alloc((size_t)M*di);
  run_slice(st, xz, xi, M, 2*di, 0, di);
  run_slice(st, xz, z,  M, 2*di, di, di);
  float* xc = ws.alloc((size_t)M*di);
  k_dwconv1d_silu<<<gblk((size_t)M*di),256,0,st>>>(xi,p.conv_w,p.conv_b,xc,Bb,N,di);
  float* tdt = ws.alloc((size_t)M*dtr);
  run_gemm(st, xc, p.xdt_w, nullptr, tdt, M, dtr, di, 0);
  float* delta = ws.alloc((size_t)M*di);
  run_gemm(st, tdt, p.dt_w, p.dt_b, delta, M, di, dtr, 3);   // fused softplus
  float* Bp = ws.alloc((size_t)M*DSN);
  float* Cp = ws.alloc((size_t)M*DSN);
  run_gemm(st, r, p.B_w, nullptr, Bp, M, DSN, d, 0);
  run_gemm(st, i, p.C_w, nullptr, Cp, M, DSN, d, 0);
  float* yv = ws.alloc((size_t)M*di);
  k_ssm_scan<<<dim3((di+127)/128,(unsigned)Bb),128,0,st>>>(xc,delta,Bp,Cp,p.A_log,p.D,yv,Bb,N,di);
  float* gt = ws.alloc((size_t)M*di);
  k_gate_mul<<<gblk((size_t)M*di),256,0,st>>>(yv,z,gt,(size_t)M*di);
  run_gemm(st, gt, p.out_w, nullptr, out, M, d, di, 0);
  ws.off = mark;
}

static void run_token_swap(hipStream_t st, WS& ws, const PTS& p, float* r, float* i,
                           int Bb, int N, int C){
  int M = Bb*N; size_t tot = (size_t)M*C;
  size_t mark = ws.off;
  float* rn  = ws.alloc(tot); float* inn = ws.alloc(tot);
  run_ln(st, r, p.ln1_g, p.ln1_b, rn, M, C);
  run_ln(st, i, p.ln2_g, p.ln2_b, inn, M, C);
  float* rsw = ws.alloc(tot); float* isw = ws.alloc(tot);
  k_swap_half<<<gblk(tot),256,0,st>>>(rn,inn,rsw,isw,tot,C);
  float* rm = ws.alloc(tot); float* im = ws.alloc(tot);
  run_mamba(st, ws, p.m_rgb, rsw, rm, Bb, N, C);
  run_mamba(st, ws, p.m_ir,  isw, im, Bb, N, C);
  k_add<<<gblk(tot),256,0,st>>>(rm, r, rm, tot);
  k_add<<<gblk(tot),256,0,st>>>(im, i, im, tot);
  int red = (C<100) ? 4 : 16, H = C/red;
  k_se<<<(unsigned)Bb, (unsigned)C, (C+H)*sizeof(float), st>>>(rm,p.se_rgb.w1,p.se_rgb.b1,p.se_rgb.w2,p.se_rgb.b2,r,N,C,H);
  k_se<<<(unsigned)Bb, (unsigned)C, (C+H)*sizeof(float), st>>>(im,p.se_ir.w1,p.se_ir.b1,p.se_ir.w2,p.se_ir.b2,i,N,C,H);
  ws.off = mark;
}

static void run_level(hipStream_t st, WS& ws, const PLvl& p, const float* rgb,
                      const float* ir, float* out, int Bb, int h, int C){
  int N = h*h, M = Bb*N; size_t tot = (size_t)M*C;
  float* tr = ws.alloc(tot); float* ti = ws.alloc(tot);
  k_copy<<<gblk(tot),256,0,st>>>(rgb, tr, tot);
  k_copy<<<gblk(tot),256,0,st>>>(ir,  ti, tot);
  run_token_swap(st, ws, p.ts1, tr, ti, Bb, N, C);
  run_token_swap(st, ws, p.ts2, tr, ti, Bb, N, C);
  float* Ir = ws.alloc(tot);
  k_conv3x3<<<gblk(tot),256,0,st>>>(tr,ti,p.sf1_w,p.sf1_b,tr,Ir,Bb,h,h,C,C,C,0);
  float* Ii = ws.alloc(tot);
  k_conv3x3<<<gblk(tot),256,0,st>>>(ti,Ir,p.sf2_w,p.sf2_b,ti,Ii,Bb,h,h,C,C,C,1);
  float* att = ws.alloc((size_t)M*2);
  k_conv3x3<<<gblk((size_t)M*2),256,0,st>>>(Ir,Ii,p.la_w,p.la_b,nullptr,att,Bb,h,h,C,C,2,0);
  float* ff = ws.alloc(tot);
  k_softmax2_blend<<<gblk(tot),256,0,st>>>(att,Ir,Ii,ff,(size_t)M,C);
  float* skip = ws.alloc(tot);
  k_conv1x1<<<gblk(tot),256,0,st>>>(ff,p.skip_w,p.skip_b,skip,(size_t)M,C);
  float* trn = ws.alloc(tot); float* tin = ws.alloc(tot); float* tf = ws.alloc(tot);
  run_ln(st, Ir, p.pe_g, p.pe_b, trn, M, C);
  run_ln(st, Ii, p.pe_g, p.pe_b, tin, M, C);
  run_ln(st, ff, p.pe_g, p.pe_b, tf,  M, C);
  float* res = ws.alloc(tot);
  k_fill<<<gblk(tot),256,0,st>>>(res, 0.f, tot);
  float* fcur = tf; float* rescur = res;
  for(int j=0;j<3;j++){
    const PM3& m = p.m3[j];
    float* fnew = ws.alloc(tot);   // persists: next iteration's res
    float* fout = ws.alloc(tot);   // persists: next iteration's f
    size_t m2 = ws.off;
    k_add<<<gblk(tot),256,0,st>>>(fcur, rescur, fnew, tot);
    float* fn  = ws.alloc(tot);
    float* rn2 = ws.alloc(tot);
    float* in2 = ws.alloc(tot);
    run_ln(st, fnew, m.ln0_g, m.ln0_b, fn,  M, C);
    run_ln(st, trn,  m.ln1_g, m.ln1_b, rn2, M, C);
    run_ln(st, tin,  m.ln2_g, m.ln2_b, in2, M, C);
    float* yv = ws.alloc(tot);
    run_cross(st, ws, m.mamba, fn, rn2, in2, yv, Bb, N, C);
    k_dwconv3x3<<<gblk(tot),256,0,st>>>(yv, m.dw_w, m.dw_b, fout, Bb, h, h, C);
    ws.off = m2;
    fcur = fout; rescur = fnew;
  }
  k_add<<<gblk(tot),256,0,st>>>(fcur, skip, out, tot);
}

extern "C" void kernel_launch(void* const* d_in, const int* in_sizes, int n_in,
                              void* d_out, int out_size, void* d_ws, size_t ws_size,
                              hipStream_t stream){
  (void)in_sizes; (void)n_in; (void)out_size;
  // setup_inputs() insertion order: rgb0, ir0, rgb1, ir1, ..., then params leaves.
  const float* rgb[4]; const float* ir[4];
  for(int l=0;l<4;l++){ rgb[l] = (const float*)d_in[2*l]; ir[l] = (const float*)d_in[2*l+1]; }
  Cur cur{d_in, 8};
  PLvl lv[4];
  for(int l=0;l<4;l++) load_level(cur, lv[l]);
  float* out = (float*)d_out;
  size_t ooff = 0;
  WS ws{(char*)d_ws, 0, ws_size};
  for(int l=0;l<4;l++){
    ws.off = 0;                       // levels are independent: reuse scratch
    int h = 128 >> l, C = 16 << l, N = h*h;
    run_level(stream, ws, lv[l], rgb[l], ir[l], out + ooff, 2, h, C);
    ooff += (size_t)2*N*C;
  }
}
